// GAT_50989851738451
// MI455X (gfx1250) — matrix operand order repaired and checked by reading, not compiled, not run on hardware
//
#include <hip/hip_runtime.h>
#include <hip/hip_bf16.h>
#include <math.h>
#include <stdint.h>

// ---------------- problem constants (match reference) ----------------
#define Nn   100000
#define Ee   1600000
#define DF   128          // d_cat = HEADS*HID = 128 (also F_IN)
#define NCLS 8
#define SLOPE 0.2f
#define EPSn  1e-16f
#define ROWS 80           // rows per transform block: 100000 = 80 * 1250

typedef __attribute__((ext_vector_type(16))) _Float16 v16h;
typedef __attribute__((ext_vector_type(8)))  float    v8f;
typedef __attribute__((ext_vector_type(4)))  uint32_t u32x4;
typedef __attribute__((ext_vector_type(8)))  uint32_t u32x8;

// ---------------- helpers ----------------
__device__ __forceinline__ float leaky(float x) { return x > 0.0f ? x : SLOPE * x; }

// float atomic max via int/uint trick (valid for init = -FLT_MAX)
__device__ __forceinline__ void atomicMaxF(float* addr, float val) {
  if (val >= 0.0f) atomicMax((int*)addr, __float_as_int(val));
  else             atomicMin((unsigned int*)addr, __float_as_uint(val));
}

// ---------------- generic fill ----------------
__global__ void k_fill(float* __restrict__ p, float v, int n) {
  int i = blockIdx.x * blockDim.x + threadIdx.x;
  if (i < n) p[i] = v;
}

// ---------------- W pre-convert: W (128x128 f32) -> W^T (128x128 f16) ---
__global__ void k_convW(const float* __restrict__ W, _Float16* __restrict__ WhT) {
  int i = blockIdx.x * blockDim.x + threadIdx.x;
  if (i >= DF * DF) return;
  int k = i >> 7, c = i & 127;
  WhT[c * DF + k] = (_Float16)W[i];
}

// ---------------- WMMA feature transform: H = X (Nx128) * W (128x128) ---
// One block = 80 rows (5 M-tiles); 8 waves each own a 16-wide column slab.
// W^T (pre-converted f16) is DMAed into LDS by the Tensor Data Mover with
// hardware padding (4 dwords every 64 dwords -> row stride 136 f16) to kill
// bank conflicts; A tile converted f32->f16 cooperatively. Each wave hoists
// all 4 B fragments and runs 5x4 = 20 v_wmma_f32_16x16x32_f16 ops.
#define AHS (DF + 8)   // padded A row stride (f16 elems)
#define WTS (DF + 8)   // padded W^T row stride (f16 elems)

__global__ __launch_bounds__(256) void k_transform_wmma(
    const float* __restrict__ X, const _Float16* __restrict__ WhT,
    float* __restrict__ Hout) {
  __shared__ _Float16 WT[DF * WTS];      // 34816 B, filled by TDM (padded)
  __shared__ _Float16 Ah[ROWS * AHS];    // 21760 B
  const int t    = threadIdx.x;
  const int lane = t & 31;
  const int wv   = t >> 5;               // 8 waves per block (wave32)
  const int row0 = blockIdx.x * ROWS;

  if (wv == 0) {
    // --- Tensor DMA: 16384 f16 linear, pad 4 dwords per 64 dwords ---
    uint64_t ga = (uint64_t)(uintptr_t)WhT;
    uint32_t la = (uint32_t)(uintptr_t)(&WT[0]);
    u32x4 g0;
    g0[0] = 1u;                                   // count=1, user descriptor
    g0[1] = la;                                   // lds_addr
    g0[2] = (uint32_t)ga;                         // global_addr[31:0]
    g0[3] = (uint32_t)(ga >> 32) | (2u << 30);    // global_addr[56:32] | type=2
    u32x8 g1;
    g1[0] = (1u << 16)          // data_size = 2 bytes
          | (1u << 20)          // pad_enable
          | (5u << 22)          // pad_interval: 64 dwords
          | (3u << 25);         // pad_amount: 4 dwords
    g1[1] = (16384u & 0xFFFFu) << 16;             // tensor_dim0[15:0]
    g1[2] = (16384u >> 16) | (1u << 16);          // dim0[31:16] | dim1[15:0]=1
    g1[3] = (16384u << 16);                       // tile_dim0 = 16384
    g1[4] = 0u;                                   // tile_dim1/2 unused
    g1[5] = 16384u;                               // tensor_dim0_stride
    g1[6] = 0u;
    g1[7] = 0u;
    u32x4 gz;
    gz[0] = 0u; gz[1] = 0u; gz[2] = 0u; gz[3] = 0u;
    asm volatile("tensor_load_to_lds %0, %1, %2, %3"
                 :: "s"(g0), "s"(g1), "s"(gz), "s"(gz)
                 : "memory");
    __builtin_amdgcn_s_wait_tensorcnt(0);
  }

  // prefetch next block's A tile toward L2 (global_prefetch_b8 path)
  if (row0 + ROWS < Nn && t < 32)
    __builtin_prefetch(X + (size_t)(row0 + ROWS) * DF + t * 8, 0, 1);

  // cooperative A tile load (coalesced) with f32->f16 convert
  for (int i = t; i < ROWS * DF; i += 256) {
    int r = i >> 7, k = i & 127;
    Ah[r * AHS + k] = (_Float16)X[(size_t)(row0 + r) * DF + k];
  }
  __syncthreads();

  const int mn   = lane & 15;
  const int g    = lane >> 4;
  const int col0 = wv * 16;

  // hoist all B fragments (K = 0..127): elem e -> K = kk*32 + 16g + e
  v16h bf[4];
  #pragma unroll
  for (int kk = 0; kk < 4; ++kk) {
    #pragma unroll
    for (int e = 0; e < 16; ++e)
      bf[kk][e] = WT[(col0 + mn) * WTS + kk * 32 + 16 * g + e];
  }

  #pragma unroll
  for (int mt = 0; mt < ROWS / 16; ++mt) {
    v8f acc = {};
    const int rb = mt * 16;
    #pragma unroll
    for (int kk = 0; kk < 4; ++kk) {
      v16h a;
      #pragma unroll
      for (int j = 0; j < 8; ++j) {
        a[j]     = Ah[(rb + mn) * AHS + kk * 32 + 16 * g + j];
        a[8 + j] = Ah[(rb + mn) * AHS + kk * 32 + 16 * g + 8 + j];
      }
      acc = __builtin_amdgcn_wmma_f32_16x16x32_f16(
          /*neg_a=*/false, a, /*neg_b=*/false, bf[kk],
          /*c_mod=*/(short)0, acc, /*reuse_a=*/false, /*reuse_b=*/false);
    }
    // C/D layout: VGPR v -> row m = v + 8*g, col = mn
    const int orow = row0 + rb + 8 * g;
    #pragma unroll
    for (int v = 0; v < 8; ++v)
      Hout[(size_t)(orow + v) * DF + col0 + mn] = acc[v];
  }
}

// ---------------- layer-3 small GEMM: H3 = X (Nx128) * W3 (128x8) ------
__global__ void k_gemm8(const float* __restrict__ X, const float* __restrict__ W,
                        float* __restrict__ Hout) {
  int idx = blockIdx.x * blockDim.x + threadIdx.x;
  if (idx >= Nn * NCLS) return;
  int n = idx >> 3, c = idx & 7;
  const float* xp = X + (size_t)n * DF;
  float s = 0.0f;
  #pragma unroll 8
  for (int k = 0; k < DF; ++k) s += xp[k] * W[k * NCLS + c];
  Hout[idx] = s;
}

// ---------------- per-node attention scores a_src/a_dst ----------------
__global__ void k_att(const float* __restrict__ Hf, const float* __restrict__ As,
                      const float* __restrict__ Ad, float* __restrict__ a_s,
                      float* __restrict__ a_d, int H, int C) {
  int idx = blockIdx.x * blockDim.x + threadIdx.x;
  if (idx >= Nn * H) return;
  int n = idx / H, h = idx % H;
  const float* hp = Hf + (size_t)n * (H * C) + h * C;
  float ss = 0.0f, sd = 0.0f;
  for (int c = 0; c < C; ++c) {
    float v = hp[c];
    ss += v * As[h * C + c];
    sd += v * Ad[h * C + c];
  }
  a_s[idx] = ss;
  a_d[idx] = sd;
}

// ---------------- edge pass 1: segment max of leaky logits -------------
__global__ void k_edge_max(const int* __restrict__ src, const int* __restrict__ dst,
                           const float* __restrict__ a_s, const float* __restrict__ a_d,
                           float* __restrict__ emx, int Etot, int H) {
  int idx = blockIdx.x * blockDim.x + threadIdx.x;
  if (idx >= Etot * H) return;
  int e = idx / H, h = idx % H;
  int s = (e < Ee) ? src[e] : (e - Ee);  // self-loops appended
  int d = (e < Ee) ? dst[e] : (e - Ee);
  atomicMaxF(&emx[d * H + h], leaky(a_s[s * H + h] + a_d[d * H + h]));
}

// ---------------- edge pass 2: segment sum of exp ----------------------
__global__ void k_edge_sum(const int* __restrict__ src, const int* __restrict__ dst,
                           const float* __restrict__ a_s, const float* __restrict__ a_d,
                           const float* __restrict__ emx, float* __restrict__ den,
                           int Etot, int H) {
  int idx = blockIdx.x * blockDim.x + threadIdx.x;
  if (idx >= Etot * H) return;
  int e = idx / H, h = idx % H;
  int s = (e < Ee) ? src[e] : (e - Ee);
  int d = (e < Ee) ? dst[e] : (e - Ee);
  float lg = leaky(a_s[s * H + h] + a_d[d * H + h]);
  atomicAdd(&den[d * H + h], expf(lg - emx[d * H + h]));
}

// ---------------- edge pass 3: weighted scatter of h[src] --------------
// one thread = one edge x 4 channels (chunk stays within one head: C % 4 == 0)
__global__ void k_edge_msg(const int* __restrict__ src, const int* __restrict__ dst,
                           const float* __restrict__ Hf, const float* __restrict__ a_s,
                           const float* __restrict__ a_d, const float* __restrict__ emx,
                           const float* __restrict__ den, float* __restrict__ outb,
                           int Etot, int H, int C) {
  const int dfeat  = H * C;
  const int chunks = dfeat >> 2;
  int idx = blockIdx.x * blockDim.x + threadIdx.x;
  if (idx >= Etot * chunks) return;
  int e = idx / chunks, q = idx % chunks;
  int s = (e < Ee) ? src[e] : (e - Ee);
  int d = (e < Ee) ? dst[e] : (e - Ee);
  int h = (q << 2) / C;
  float lg = leaky(a_s[s * H + h] + a_d[d * H + h]);
  float al = expf(lg - emx[d * H + h]) / (den[d * H + h] + EPSn);
  const float4 hv = *(const float4*)(Hf + (size_t)s * dfeat + (q << 2));
  float* op = outb + (size_t)d * dfeat + (q << 2);
  atomicAdd(op + 0, hv.x * al);
  atomicAdd(op + 1, hv.y * al);
  atomicAdd(op + 2, hv.z * al);
  atomicAdd(op + 3, hv.w * al);
}

// ---------------- bias + ELU (layers 1,2) ------------------------------
__global__ void k_bias_elu(float* __restrict__ buf, const float* __restrict__ bias,
                           int n, int dfeat) {
  int i = blockIdx.x * blockDim.x + threadIdx.x;
  if (i >= n) return;
  float v = buf[i] + bias[i % dfeat];
  buf[i] = v > 0.0f ? v : (expf(v) - 1.0f);
}

// ---------------- final: bias + log_softmax over 8 classes -------------
__global__ void k_final(const float* __restrict__ acc, const float* __restrict__ bias,
                        float* __restrict__ out) {
  int n = blockIdx.x * blockDim.x + threadIdx.x;
  if (n >= Nn) return;
  float v[NCLS], m = -3.0e38f;
  #pragma unroll
  for (int c = 0; c < NCLS; ++c) { v[c] = acc[n * NCLS + c] + bias[c]; m = fmaxf(m, v[c]); }
  float s = 0.0f;
  #pragma unroll
  for (int c = 0; c < NCLS; ++c) s += expf(v[c] - m);
  float ls = logf(s);
  #pragma unroll
  for (int c = 0; c < NCLS; ++c) out[n * NCLS + c] = v[c] - m - ls;
}

// ---------------- host orchestration -----------------------------------
extern "C" void kernel_launch(void* const* d_in, const int* in_sizes, int n_in,
                              void* d_out, int out_size, void* d_ws, size_t ws_size,
                              hipStream_t stream) {
  const float* x   = (const float*)d_in[0];
  const int*   ei  = (const int*)  d_in[1];
  const float* W1  = (const float*)d_in[2];
  const float* as1 = (const float*)d_in[3];
  const float* ad1 = (const float*)d_in[4];
  const float* b1  = (const float*)d_in[5];
  const float* W2  = (const float*)d_in[6];
  const float* as2 = (const float*)d_in[7];
  const float* ad2 = (const float*)d_in[8];
  const float* b2  = (const float*)d_in[9];
  const float* W3  = (const float*)d_in[10];
  const float* as3 = (const float*)d_in[11];
  const float* ad3 = (const float*)d_in[12];
  const float* b3  = (const float*)d_in[13];

  float* bufA = (float*)d_ws;                    // transformed features h (51.2 MB)
  float* bufB = bufA + (size_t)Nn * DF;          // layer i/o (51.2 MB)
  float* a_s  = bufB + (size_t)Nn * DF;          // [N,H]
  float* a_d  = a_s + (size_t)Nn * 4;
  float* emx  = a_d + (size_t)Nn * 4;
  float* den  = emx + (size_t)Nn * 4;
  float* h3   = den + (size_t)Nn * 4;            // [N,8]
  float* acc3 = h3 + (size_t)Nn * NCLS;          // [N,8]
  _Float16* whT = (_Float16*)(acc3 + (size_t)Nn * NCLS);  // 128x128 f16 (32 KB)

  const int Etot = Ee + Nn;
  const int* srcp = ei;
  const int* dstp = ei + Ee;

  auto blocks = [](int n) { return dim3((unsigned)((n + 255) / 256)); };
  const dim3 TB(256);

  // ---------------- layer 1 (in: x) ----------------
  k_convW<<<blocks(DF * DF), TB, 0, stream>>>(W1, whT);
  k_transform_wmma<<<dim3(Nn / ROWS), TB, 0, stream>>>(x, whT, bufA);
  k_att<<<blocks(Nn * 4), TB, 0, stream>>>(bufA, as1, ad1, a_s, a_d, 4, 32);
  k_fill<<<blocks(Nn * 4), TB, 0, stream>>>(emx, -3.0e38f, Nn * 4);
  k_fill<<<blocks(Nn * 4), TB, 0, stream>>>(den, 0.0f, Nn * 4);
  k_fill<<<blocks(Nn * DF), TB, 0, stream>>>(bufB, 0.0f, Nn * DF);
  k_edge_max<<<blocks(Etot * 4), TB, 0, stream>>>(srcp, dstp, a_s, a_d, emx, Etot, 4);
  k_edge_sum<<<blocks(Etot * 4), TB, 0, stream>>>(srcp, dstp, a_s, a_d, emx, den, Etot, 4);
  k_edge_msg<<<blocks(Etot * 32), TB, 0, stream>>>(srcp, dstp, bufA, a_s, a_d, emx, den,
                                                   bufB, Etot, 4, 32);
  k_bias_elu<<<blocks(Nn * DF), TB, 0, stream>>>(bufB, b1, Nn * DF, DF);

  // ---------------- layer 2 (in: bufB) ----------------
  k_convW<<<blocks(DF * DF), TB, 0, stream>>>(W2, whT);
  k_transform_wmma<<<dim3(Nn / ROWS), TB, 0, stream>>>(bufB, whT, bufA);
  k_att<<<blocks(Nn * 4), TB, 0, stream>>>(bufA, as2, ad2, a_s, a_d, 4, 32);
  k_fill<<<blocks(Nn * 4), TB, 0, stream>>>(emx, -3.0e38f, Nn * 4);
  k_fill<<<blocks(Nn * 4), TB, 0, stream>>>(den, 0.0f, Nn * 4);
  k_fill<<<blocks(Nn * DF), TB, 0, stream>>>(bufB, 0.0f, Nn * DF);  // GEMM already consumed it
  k_edge_max<<<blocks(Etot * 4), TB, 0, stream>>>(srcp, dstp, a_s, a_d, emx, Etot, 4);
  k_edge_sum<<<blocks(Etot * 4), TB, 0, stream>>>(srcp, dstp, a_s, a_d, emx, den, Etot, 4);
  k_edge_msg<<<blocks(Etot * 32), TB, 0, stream>>>(srcp, dstp, bufA, a_s, a_d, emx, den,
                                                   bufB, Etot, 4, 32);
  k_bias_elu<<<blocks(Nn * DF), TB, 0, stream>>>(bufB, b2, Nn * DF, DF);

  // ---------------- layer 3 (in: bufB, 1 head, 8 ch) ----------------
  k_gemm8<<<blocks(Nn * NCLS), TB, 0, stream>>>(bufB, W3, h3);
  k_att<<<blocks(Nn), TB, 0, stream>>>(h3, as3, ad3, a_s, a_d, 1, NCLS);
  k_fill<<<blocks(Nn), TB, 0, stream>>>(emx, -3.0e38f, Nn);
  k_fill<<<blocks(Nn), TB, 0, stream>>>(den, 0.0f, Nn);
  k_fill<<<blocks(Nn * NCLS), TB, 0, stream>>>(acc3, 0.0f, Nn * NCLS);
  k_edge_max<<<blocks(Etot), TB, 0, stream>>>(srcp, dstp, a_s, a_d, emx, Etot, 1);
  k_edge_sum<<<blocks(Etot), TB, 0, stream>>>(srcp, dstp, a_s, a_d, emx, den, Etot, 1);
  k_edge_msg<<<blocks(Etot * 2), TB, 0, stream>>>(srcp, dstp, h3, a_s, a_d, emx, den,
                                                  acc3, Etot, 1, NCLS);
  k_final<<<blocks(Nn), TB, 0, stream>>>(acc3, b3, (float*)d_out);
}
